// Features_70806830841878
// MI455X (gfx1250) — matrix operand order repaired and checked by reading, not compiled, not run on hardware
//
#include <hip/hip_runtime.h>
#include <math.h>

typedef __bf16 bf16_t;
typedef __attribute__((ext_vector_type(16))) __bf16 v16bf;
typedef __attribute__((ext_vector_type(8)))  float  v8f;

#define N_PATCH 3136
#define M_LIB   40000
#define HWS     56
#define IMG     224
#define IMGPIX  (IMG*IMG)
#define MSPLIT  20
#define FLT_BIG 3.402823466e38f

union ABFrag { uint4 u4[2]; v16bf v; };

__device__ inline bf16_t f2bf(float f) { return (bf16_t)f; }

// ---------------------------------------------------------------------------
// Normalize (optional) + convert to bf16, and compute squared row norms of the
// bf16-rounded values (so d2 = a2 + b2 - 2ab is consistent in bf16 space).
// ---------------------------------------------------------------------------
__global__ __launch_bounds__(256)
void prep_kernel(const float* __restrict__ src, const float* __restrict__ mean,
                 const float* __restrict__ stdv, bf16_t* __restrict__ dstbf,
                 float* __restrict__ nrm2, int D, int doNorm)
{
    __shared__ float red[256];
    const int row = blockIdx.x;
    const int tid = threadIdx.x;
    float mu = 0.f, inv = 1.f;
    if (doNorm) { mu = mean[0]; inv = 1.0f / stdv[0]; }
    const float* s = src + (size_t)row * D;
    bf16_t* d = dstbf + (size_t)row * D;
    float acc = 0.f;
    for (int i = tid; i < D; i += 256) {
        float v = (s[i] - mu) * inv;
        bf16_t b = f2bf(v);
        d[i] = b;
        float vr = (float)b;
        acc += vr * vr;
    }
    red[tid] = acc;
    __syncthreads();
    for (int o = 128; o > 0; o >>= 1) {
        if (tid < o) red[tid] += red[tid + o];
        __syncthreads();
    }
    if (tid == 0) nrm2[row] = red[0];
}

// ---------------------------------------------------------------------------
// Fused GEMM + row-min. Each wave owns TWO 16-col lib tiles sharing one A
// fragment; the k-loop is software-pipelined (double-buffered fragments) so
// next-step b128 loads overlap the current v_wmma_f32_16x16x32_bf16 pair.
// The A-tile LDS offset is made opaque per tile-iteration (asm barrier on a
// 32-bit offset) so LICM cannot hoist all k-steps' A fragments into registers
// (which previously caused either >256 VGPRs or scratch spills).
// The hot loop tracks min of d^2 (sqrt is monotone -> applied later).
// ---------------------------------------------------------------------------
template<int D>
__global__ __launch_bounds__(256)
void dist_min_kernel(const bf16_t* __restrict__ patchbf, const bf16_t* __restrict__ libbf,
                     const float* __restrict__ a2, const float* __restrict__ b2,
                     float* __restrict__ pval, int* __restrict__ pidx, int N, int M)
{
    __shared__ alignas(16) bf16_t sA[16 * D];
    __shared__ float sA2[16];
    __shared__ float sPv[128];
    __shared__ int   sPi[128];

    const int tid  = threadIdx.x;
    const int wave = tid >> 5;
    const int lane = tid & 31;
    const int half = lane >> 4;
    const int l15  = lane & 15;
    const int n0   = blockIdx.x * 16;

    // Stage the 16 x D bf16 A tile into LDS (contiguous rows).
    {
        const uint4* s = (const uint4*)(patchbf + (size_t)n0 * D);
        uint4* d = (uint4*)sA;
        const int nvec = (16 * D * (int)sizeof(bf16_t)) / 16;
        for (int i = tid; i < nvec; i += 256) d[i] = s[i];
    }
    if (tid < 16) sA2[tid] = a2[n0 + tid];
    __syncthreads();

    const int tiles   = M >> 4;
    const int tps     = (tiles + gridDim.y - 1) / gridDim.y;
    const int tile0   = blockIdx.y * tps;
    const int tileEnd = min(tile0 + tps, tiles);

    // Row norms for this lane's 8 C-rows, hoisted out of the tile loop.
    float ra2[8];
#pragma unroll
    for (int g = 0; g < 8; ++g) ra2[g] = sA2[g + half * 8];

    float bestv[8]; int besti[8];
#pragma unroll
    for (int g = 0; g < 8; ++g) { bestv[g] = FLT_BIG; besti[g] = 0; }

    // A fragment per lane: row = l15, K contiguous 16 starting at 16*half.
    const unsigned aOffBase = (unsigned)(l15 * D + half * 16);

    for (int t0 = tile0 + wave * 2; t0 < tileEnd; t0 += 16) {
        const int t1 = t0 + 1;
        const bool has1 = (t1 < tileEnd);
        const int t1c = has1 ? t1 : t0;   // clamp: loads stay in-bounds
        const int c0 = t0 << 4;
        const int c1 = t1c << 4;

        // Opaque 32-bit LDS offset: stays a ds_load base, defeats cross-tile
        // hoisting/CSE of the A-fragment loads.
        unsigned aOff = aOffBase;
        asm volatile("" : "+v"(aOff));
        const bf16_t* aPtr = sA + aOff;

        // B fragment per lane: col = l15, K contiguous 16 starting at 16*half.
        const bf16_t* b0p = libbf + (size_t)(c0 + l15) * D + half * 16;
        const bf16_t* b1p = libbf + (size_t)(c1 + l15) * D + half * 16;

        v8f acc0 = {0.f, 0.f, 0.f, 0.f, 0.f, 0.f, 0.f, 0.f};
        v8f acc1 = {0.f, 0.f, 0.f, 0.f, 0.f, 0.f, 0.f, 0.f};

        ABFrag fa, fb0, fb1;
        fa.u4[0]  = *(const uint4*)(aPtr);
        fa.u4[1]  = *(const uint4*)(aPtr + 8);
        fb0.u4[0] = *(const uint4*)(b0p);
        fb0.u4[1] = *(const uint4*)(b0p + 8);
        fb1.u4[0] = *(const uint4*)(b1p);
        fb1.u4[1] = *(const uint4*)(b1p + 8);

#pragma unroll 2
        for (int k0 = 32; k0 < D; k0 += 32) {
            ABFrag na, nb0, nb1;
            na.u4[0]  = *(const uint4*)(aPtr + k0);
            na.u4[1]  = *(const uint4*)(aPtr + k0 + 8);
            nb0.u4[0] = *(const uint4*)(b0p + k0);
            nb0.u4[1] = *(const uint4*)(b0p + k0 + 8);
            nb1.u4[0] = *(const uint4*)(b1p + k0);
            nb1.u4[1] = *(const uint4*)(b1p + k0 + 8);
            acc0 = __builtin_amdgcn_wmma_f32_16x16x32_bf16(false, fa.v, false, fb0.v,
                                                           (short)0, acc0, false, false);
            acc1 = __builtin_amdgcn_wmma_f32_16x16x32_bf16(false, fa.v, false, fb1.v,
                                                           (short)0, acc1, false, false);
            fa = na; fb0 = nb0; fb1 = nb1;
        }
        acc0 = __builtin_amdgcn_wmma_f32_16x16x32_bf16(false, fa.v, false, fb0.v,
                                                       (short)0, acc0, false, false);
        acc1 = __builtin_amdgcn_wmma_f32_16x16x32_bf16(false, fa.v, false, fb1.v,
                                                       (short)0, acc1, false, false);

        // Epilogue: fold min of d^2 (no sqrt in the hot loop).
        const float b2v0 = b2[c0 + l15];
        const int   idx0 = c0 + l15;
#pragma unroll
        for (int g = 0; g < 8; ++g) {
            float d2 = fmaf(-2.0f, acc0[g], ra2[g] + b2v0);
            if (d2 < bestv[g]) { bestv[g] = d2; besti[g] = idx0; }
        }
        if (has1) {
            const float b2v1 = b2[c1 + l15];
            const int   idx1 = c1 + l15;
#pragma unroll
            for (int g = 0; g < 8; ++g) {
                float d2 = fmaf(-2.0f, acc1[g], ra2[g] + b2v1);
                if (d2 < bestv[g]) { bestv[g] = d2; besti[g] = idx1; }
            }
        }
    }

    // Reduce across the 16 lanes holding different columns of the same row.
#pragma unroll
    for (int g = 0; g < 8; ++g) {
#pragma unroll
        for (int off = 1; off < 16; off <<= 1) {
            float ov = __shfl_xor(bestv[g], off, 32);
            int   oi = __shfl_xor(besti[g], off, 32);
            if (ov < bestv[g] || (ov == bestv[g] && oi < besti[g])) { bestv[g] = ov; besti[g] = oi; }
        }
    }
    if (l15 == 0) {
#pragma unroll
        for (int g = 0; g < 8; ++g) {
            int r = g + half * 8;
            sPv[wave * 16 + r] = bestv[g];
            sPi[wave * 16 + r] = besti[g];
        }
    }
    __syncthreads();
    if (tid < 16) {
        float bv = sPv[tid]; int bi = sPi[tid];
        for (int w = 1; w < 8; ++w) {
            float ov = sPv[w * 16 + tid]; int oi = sPi[w * 16 + tid];
            if (ov < bv || (ov == bv && oi < bi)) { bv = ov; bi = oi; }
        }
        size_t o = (size_t)blockIdx.y * N + n0 + tid;
        pval[o] = bv; pidx[o] = bi;   // bv is min d^2 for this row/slice
    }
}

// Partials hold min d^2; apply sqrt(max(.,eps)) once here.
__global__ __launch_bounds__(256)
void reduce_partials(const float* __restrict__ pval, const int* __restrict__ pidx,
                     float* __restrict__ minval, int* __restrict__ minidx, int N, int S)
{
    const int n = blockIdx.x * 256 + threadIdx.x;
    if (n >= N) return;
    float bv = pval[n]; int bi = pidx[n];
    for (int s = 1; s < S; ++s) {
        float ov = pval[(size_t)s * N + n]; int oi = pidx[(size_t)s * N + n];
        if (ov < bv || (ov == bv && oi < bi)) { bv = ov; bi = oi; }
    }
    minval[n] = sqrtf(fmaxf(bv, 1e-12f));
    minidx[n] = bi;
}

// scal layout: [0]=s_star(f32) [1]=s_idx(int) [2]=m_star_idx(int)
//              [3]=nn1(int) [4]=nn2(int) [5]=s (f32 final score)
__global__ __launch_bounds__(256)
void argmax_kernel(const float* __restrict__ minval, const int* __restrict__ minidx,
                   float* __restrict__ scal, int N)
{
    __shared__ float rv[256]; __shared__ int ri[256];
    const int tid = threadIdx.x;
    float bv = -FLT_BIG; int bi = 0x7fffffff;
    for (int n = tid; n < N; n += 256) {
        float v = minval[n];
        if (v > bv || (v == bv && n < bi)) { bv = v; bi = n; }
    }
    rv[tid] = bv; ri[tid] = bi;
    __syncthreads();
    for (int o = 128; o > 0; o >>= 1) {
        if (tid < o) {
            float ov = rv[tid + o]; int oi = ri[tid + o];
            if (ov > rv[tid] || (ov == rv[tid] && oi < ri[tid])) { rv[tid] = ov; ri[tid] = oi; }
        }
        __syncthreads();
    }
    if (tid == 0) {
        scal[0] = rv[0];
        ((int*)scal)[1] = ri[0];
        ((int*)scal)[2] = minidx[ri[0]];
    }
}

__global__ __launch_bounds__(256)
void wdist_kernel(const float* __restrict__ lib, const float* __restrict__ scal,
                  float* __restrict__ wdist, int M, int D)
{
    const int m = blockIdx.x * 8 + (threadIdx.x >> 5);
    const int lane = threadIdx.x & 31;
    if (m >= M) return;
    const int mstar = ((const int*)scal)[2];
    const float* a = lib + (size_t)mstar * D;
    const float* b = lib + (size_t)m * D;
    float s = 0.f;
    for (int d = lane; d < D; d += 32) { float t = a[d] - b[d]; s += t * t; }
#pragma unroll
    for (int off = 16; off > 0; off >>= 1) s += __shfl_xor(s, off, 32);
    if (lane == 0) wdist[m] = sqrtf(fmaxf(s, 1e-12f));
}

__device__ inline void top3_insert(float v, int m, float* tv, int* ti)
{
    if (v < tv[0] || (v == tv[0] && m < ti[0])) {
        tv[2] = tv[1]; ti[2] = ti[1]; tv[1] = tv[0]; ti[1] = ti[0]; tv[0] = v; ti[0] = m;
    } else if (v < tv[1] || (v == tv[1] && m < ti[1])) {
        tv[2] = tv[1]; ti[2] = ti[1]; tv[1] = v; ti[1] = m;
    } else if (v < tv[2] || (v == tv[2] && m < ti[2])) {
        tv[2] = v; ti[2] = m;
    }
}

__global__ __launch_bounds__(256)
void top3_kernel(const float* __restrict__ wdist, float* __restrict__ scal, int M)
{
    __shared__ float sv[256 * 3];
    __shared__ int   si[256 * 3];
    const int tid = threadIdx.x;
    float tv[3] = {FLT_BIG, FLT_BIG, FLT_BIG};
    int   ti[3] = {0x7fffffff, 0x7fffffff, 0x7fffffff};
    for (int m = tid; m < M; m += 256) top3_insert(wdist[m], m, tv, ti);
    for (int j = 0; j < 3; ++j) { sv[tid * 3 + j] = tv[j]; si[tid * 3 + j] = ti[j]; }
    __syncthreads();
    if (tid == 0) {
        float bv[3] = {FLT_BIG, FLT_BIG, FLT_BIG};
        int   bi[3] = {0x7fffffff, 0x7fffffff, 0x7fffffff};
        for (int t = 0; t < 256 * 3; ++t) top3_insert(sv[t], si[t], bv, bi);
        ((int*)scal)[3] = bi[1];   // skip bi[0] == m_star itself
        ((int*)scal)[4] = bi[2];
    }
}

__global__ __launch_bounds__(256)
void knn_score_kernel(const float* __restrict__ patch, const float* __restrict__ lib,
                      const float* __restrict__ mean, const float* __restrict__ stdv,
                      float* __restrict__ scal, int D)
{
    __shared__ float r1[256], r2[256];
    const int tid = threadIdx.x;
    const int sidx = ((const int*)scal)[1];
    const int nn1  = ((const int*)scal)[3];
    const int nn2  = ((const int*)scal)[4];
    const float mu = mean[0]; const float inv = 1.0f / stdv[0];
    const float* p  = patch + (size_t)sidx * D;
    const float* l1 = lib + (size_t)nn1 * D;
    const float* l2 = lib + (size_t)nn2 * D;
    float s1 = 0.f, s2 = 0.f;
    for (int d = tid; d < D; d += 256) {
        float mt = (p[d] - mu) * inv;
        float t1 = mt - l1[d]; s1 += t1 * t1;
        float t2 = mt - l2[d]; s2 += t2 * t2;
    }
    r1[tid] = s1; r2[tid] = s2;
    __syncthreads();
    for (int o = 128; o > 0; o >>= 1) {
        if (tid < o) { r1[tid] += r1[tid + o]; r2[tid] += r2[tid + o]; }
        __syncthreads();
    }
    if (tid == 0) {
        float d1 = sqrtf(r1[0]);
        float d2 = sqrtf(r2[0]);
        float sD = sqrtf((float)D);
        float s_star = scal[0];
        float w = 1.0f - expf(s_star / sD) / (expf(d1 / sD) + expf(d2 / sD));
        scal[5] = w * s_star;
    }
}

__global__ __launch_bounds__(256)
void resize_kernel(const float* __restrict__ src, float* __restrict__ dst)
{
    const int idx = blockIdx.x * 256 + threadIdx.x;
    if (idx >= IMGPIX) return;
    const int x = idx % IMG, y = idx / IMG;
    const float scale = (float)HWS / (float)IMG;
    float sx = ((float)x + 0.5f) * scale - 0.5f;
    float sy = ((float)y + 0.5f) * scale - 0.5f;
    int x0 = (int)floorf(sx); float fx = sx - (float)x0;
    int y0 = (int)floorf(sy); float fy = sy - (float)y0;
    int x1 = x0 + 1, y1 = y0 + 1;
    x0 = min(max(x0, 0), HWS - 1); x1 = min(max(x1, 0), HWS - 1);
    y0 = min(max(y0, 0), HWS - 1); y1 = min(max(y1, 0), HWS - 1);
    float v00 = src[y0 * HWS + x0], v01 = src[y0 * HWS + x1];
    float v10 = src[y1 * HWS + x0], v11 = src[y1 * HWS + x1];
    dst[idx] = (1.f - fy) * ((1.f - fx) * v00 + fx * v01)
             +        fy  * ((1.f - fx) * v10 + fx * v11);
}

// Zero-padded separable Gaussian blur, sigma=4 -> radius 12, 25 taps.
__global__ __launch_bounds__(256)
void blur_kernel(const float* __restrict__ src, float* __restrict__ dst, int vertical)
{
    const int idx = blockIdx.x * 256 + threadIdx.x;
    if (idx >= IMGPIX) return;
    const int x = idx % IMG, y = idx / IMG;
    float norm = 0.f;
#pragma unroll
    for (int t = -12; t <= 12; ++t) { float u = (float)t * 0.25f; norm += expf(-0.5f * u * u); }
    float acc = 0.f;
#pragma unroll
    for (int t = -12; t <= 12; ++t) {
        float u = (float)t * 0.25f;
        float wgt = expf(-0.5f * u * u);
        int xx = x, yy = y;
        if (vertical) yy = y + t; else xx = x + t;
        if (xx >= 0 && xx < IMG && yy >= 0 && yy < IMG) acc += wgt * src[yy * IMG + xx];
    }
    dst[idx] = acc / norm;
}

__global__ __launch_bounds__(256)
void final_kernel(const float* __restrict__ sxyz, const float* __restrict__ srgb,
                  const float* __restrict__ sfus,
                  const float* __restrict__ mxyz, const float* __restrict__ mrgb,
                  const float* __restrict__ mfus,
                  const float* __restrict__ dw, const float* __restrict__ doff,
                  const float* __restrict__ sw, const float* __restrict__ soff,
                  float* __restrict__ out)
{
    const int idx = blockIdx.x * 256 + threadIdx.x;
    if (idx == 0)
        out[0] = sxyz[5] * dw[0] + 0.1f * srgb[5] * dw[1] + sfus[5] * dw[2] - doff[0];
    if (idx < IMGPIX)
        out[1 + idx] = mxyz[idx] * sw[0] + 0.1f * mrgb[idx] * sw[1] + mfus[idx] * sw[2] - soff[0];
}

// ---------------------------------------------------------------------------
extern "C" void kernel_launch(void* const* d_in, const int* in_sizes, int n_in,
                              void* d_out, int out_size, void* d_ws, size_t ws_size,
                              hipStream_t stream)
{
    (void)in_sizes; (void)n_in; (void)out_size; (void)ws_size;
    const int N = N_PATCH, M = M_LIB;
    const int Ds[3] = {1152, 768, 768};
    const float* patches[3] = {(const float*)d_in[0], (const float*)d_in[1], (const float*)d_in[2]};
    const float* libs[3]    = {(const float*)d_in[3], (const float*)d_in[4], (const float*)d_in[5]};
    const float* means[3]   = {(const float*)d_in[6], (const float*)d_in[8], (const float*)d_in[10]};
    const float* stds[3]    = {(const float*)d_in[7], (const float*)d_in[9], (const float*)d_in[11]};
    const float* detect_w   = (const float*)d_in[12];
    const float* detect_off = (const float*)d_in[13];
    const float* seg_w      = (const float*)d_in[14];
    const float* seg_off    = (const float*)d_in[15];

    char* base = (char*)d_ws;
    size_t off = 0;
    auto walloc = [&](size_t bytes) -> void* {
        void* p = base + off;
        off = (off + bytes + 255) & ~(size_t)255;
        return p;
    };
    bf16_t* libbf   = (bf16_t*)walloc((size_t)M * 1152 * sizeof(bf16_t)); // ~92.2 MB, reused per modality
    bf16_t* patchbf = (bf16_t*)walloc((size_t)N * 1152 * sizeof(bf16_t)); // ~7.2 MB
    float*  a2      = (float*)walloc((size_t)N * 4);
    float*  b2      = (float*)walloc((size_t)M * 4);
    float*  pval    = (float*)walloc((size_t)MSPLIT * N * 4);
    int*    pidx    = (int*)walloc((size_t)MSPLIT * N * 4);
    float*  wdistb  = (float*)walloc((size_t)M * 4);
    float*  tmpA    = (float*)walloc((size_t)IMGPIX * 4);
    float*  tmpB    = (float*)walloc((size_t)IMGPIX * 4);
    float*  minval[3]; int* minidx[3]; float* scal[3]; float* maps[3];
    for (int i = 0; i < 3; ++i) {
        minval[i] = (float*)walloc((size_t)N * 4);
        minidx[i] = (int*)walloc((size_t)N * 4);
        scal[i]   = (float*)walloc(256);
        maps[i]   = (float*)walloc((size_t)IMGPIX * 4);
    }

    for (int mod = 0; mod < 3; ++mod) {
        const int D = Ds[mod];
        prep_kernel<<<N, 256, 0, stream>>>(patches[mod], means[mod], stds[mod], patchbf, a2, D, 1);
        prep_kernel<<<M, 256, 0, stream>>>(libs[mod], means[mod], stds[mod], libbf, b2, D, 0);
        dim3 grid(N / 16, MSPLIT);
        if (D == 1152)
            dist_min_kernel<1152><<<grid, 256, 0, stream>>>(patchbf, libbf, a2, b2, pval, pidx, N, M);
        else
            dist_min_kernel<768><<<grid, 256, 0, stream>>>(patchbf, libbf, a2, b2, pval, pidx, N, M);
        reduce_partials<<<(N + 255) / 256, 256, 0, stream>>>(pval, pidx, minval[mod], minidx[mod], N, MSPLIT);
        argmax_kernel<<<1, 256, 0, stream>>>(minval[mod], minidx[mod], scal[mod], N);
        wdist_kernel<<<(M + 7) / 8, 256, 0, stream>>>(libs[mod], scal[mod], wdistb, M, D);
        top3_kernel<<<1, 256, 0, stream>>>(wdistb, scal[mod], M);
        knn_score_kernel<<<1, 256, 0, stream>>>(patches[mod], libs[mod], means[mod], stds[mod], scal[mod], D);
        resize_kernel<<<(IMGPIX + 255) / 256, 256, 0, stream>>>(minval[mod], tmpA);
        blur_kernel<<<(IMGPIX + 255) / 256, 256, 0, stream>>>(tmpA, tmpB, 0);
        blur_kernel<<<(IMGPIX + 255) / 256, 256, 0, stream>>>(tmpB, maps[mod], 1);
    }
    final_kernel<<<(IMGPIX + 255) / 256, 256, 0, stream>>>(
        scal[0], scal[1], scal[2], maps[0], maps[1], maps[2],
        detect_w, detect_off, seg_w, seg_off, (float*)d_out);
}
